// DVSSNN_62646392979956
// MI455X (gfx1250) — matrix operand order repaired and checked by reading, not compiled, not run on hardware
//
#include <hip/hip_runtime.h>

// ---------------------------------------------------------------------------
// SNN forward for MI455X (gfx1250, wave32, WMMA).
//  Phase 1: pre1[b*T+t, h] = x[b,t,:] . w1[h,:]   (one big f16 WMMA GEMM)
//  Phase 2: persistent 17-WG kernel runs the 256 sequential LIF steps with
//           WMMA recurrent GEMMs + atomic grid barrier per step.
// ---------------------------------------------------------------------------

typedef __attribute__((ext_vector_type(16))) _Float16 v16h;
typedef __attribute__((ext_vector_type(8)))  _Float16 v8h;
typedef __attribute__((ext_vector_type(8)))  float    v8f;

namespace snn {
constexpr int kB = 64, kT = 256, kC = 2048, kH = 512, kO = 11;
constexpr int kM = kB * kT;                    // 16384 GEMM rows
constexpr float kLeak    = 0.003f;
constexpr float kOutLeak = 0.0015f;            // LEAK * 0.5
constexpr unsigned kHiddenWGs = 16;            // H = 16 slices x 32 cols
constexpr unsigned kWGs = kHiddenWGs + 1;      // + 1 output WG
}

// --------------------------- WMMA helpers ----------------------------------

__device__ __forceinline__ v8f wmma_f16(v16h a, v16h b, v8f c) {
  // v_wmma_f32_16x16x32_f16  (8-arg form: neg_a, A, neg_b, B, c_mod, C, reuse)
  return __builtin_amdgcn_wmma_f32_16x16x32_f16(false, a, false, b,
                                                (short)0, c, false, false);
}

// A fragment (16x32 f16, M x K): lane L holds row m0+(L&15).
// Lanes 0-15: halves 0-7 = K0..K0+7, halves 8-15 = K0+16..K0+23.
// Lanes 16-31: the +8 shifted runs.  (ISA 7.12.2, 16-bit A 16x32.)
__device__ __forceinline__ v16h load_frag_a(const _Float16* __restrict__ base,
                                            int ld, int m0, int k0, int lane) {
  const int row  = m0 + (lane & 15);
  const int koff = k0 + ((lane >> 4) << 3);          // +0 or +8
  const _Float16* p = base + (size_t)row * ld + koff;
  union { v16h v; v8h h[2]; } r;
  r.h[0] = *(const v8h*)(p);                          // b128
  r.h[1] = *(const v8h*)(p + 16);                     // b128
  return r.v;
}

// Same A fragment but from an f32 source, converting inline (f32->f16).
__device__ __forceinline__ v16h load_frag_a_cvt(const float* __restrict__ base,
                                                int ld, int m0, int k0, int lane) {
  const int row  = m0 + (lane & 15);
  const int koff = k0 + ((lane >> 4) << 3);
  const float* p = base + (size_t)row * ld + koff;
  v16h r;
#pragma unroll
  for (int i = 0; i < 8; ++i) r[i]     = (_Float16)p[i];
#pragma unroll
  for (int i = 0; i < 8; ++i) r[8 + i] = (_Float16)p[16 + i];
  return r;
}

// B fragment (32x16 f16, K x N) from a ROW-MAJOR weight W[N, K] (i.e. B=Wt):
// lane L holds column n0+(L&15); lanes 0-15: halves 0-7 = K0..K0+7,
// halves 8-15 = K0+16..K0+23; lanes 16-31 the +8 shifted runs (same as A).
__device__ __forceinline__ v16h load_frag_b(const _Float16* __restrict__ w,
                                            int ld, int n0, int k0, int lane) {
  const int col = n0 + (lane & 15);
  const int kk  = k0 + ((lane >> 4) << 3);           // +0 or +8
  const _Float16* p = w + (size_t)col * ld + kk;
  union { v16h v; v8h h[2]; } r;
  r.h[0] = *(const v8h*)(p);                          // b128
  r.h[1] = *(const v8h*)(p + 16);                     // b128
  return r.v;
}

// --------------------------- small prep kernels ----------------------------

__global__ void cvt_f32_to_f16(const float* __restrict__ in,
                               _Float16* __restrict__ out, int n) {
  int i = blockIdx.x * blockDim.x + threadIdx.x;
  if (i < n) out[i] = (_Float16)in[i];
}

// w2 [11,512] f32 -> w2p [16,512] f16 zero-padded (so O fits one WMMA tile).
__global__ void pad_w2_f16(const float* __restrict__ w2,
                           _Float16* __restrict__ w2p) {
  int i = blockIdx.x * blockDim.x + threadIdx.x;
  if (i < 16 * snn::kH) {
    int o = i / snn::kH, k = i % snn::kH;
    w2p[i] = (o < snn::kO) ? (_Float16)w2[o * snn::kH + k] : (_Float16)0.0f;
  }
}

// --------------------------- Phase 1: big GEMM -----------------------------
// pre1[m, h] = sum_k x2d[m,k] * w1[h,k];  m = b*T + t.
// Block = 8 waves; block m-strip of 64 rows, wave w owns columns w*64..+63.
// Per wave: 4x4 register tile of 16x16 accumulators (128 VGPRs).

__global__ __launch_bounds__(256) void snn_gemm1(
    const float* __restrict__ x, const _Float16* __restrict__ w1h,
    float* __restrict__ pre1) {
  using namespace snn;
  const int wave = threadIdx.x >> 5;
  const int lane = threadIdx.x & 31;
  const int m0 = blockIdx.x * 64;
  const int n0 = wave * 64;

  v8f acc[4][4] = {};
  for (int k0 = 0; k0 < kC; k0 += 32) {
    v16h bf[4];
#pragma unroll
    for (int ni = 0; ni < 4; ++ni)
      bf[ni] = load_frag_b(w1h, kC, n0 + ni * 16, k0, lane);
#pragma unroll
    for (int mi = 0; mi < 4; ++mi) {
      v16h af = load_frag_a_cvt(x, kC, m0 + mi * 16, k0, lane);
#pragma unroll
      for (int ni = 0; ni < 4; ++ni)
        acc[mi][ni] = wmma_f16(af, bf[ni], acc[mi][ni]);
    }
  }
  // C/D layout: VGPR r -> row m0 + (lane>=16 ? 8 : 0) + r, col n0 + (lane&15).
  const int rsel = (lane >> 4) << 3;
#pragma unroll
  for (int mi = 0; mi < 4; ++mi)
#pragma unroll
    for (int ni = 0; ni < 4; ++ni) {
      const int col   = n0 + ni * 16 + (lane & 15);
      const int rbase = m0 + mi * 16 + rsel;
#pragma unroll
      for (int r = 0; r < 8; ++r)
        pre1[(size_t)(rbase + r) * kH + col] = acc[mi][ni][r];
    }
}

// --------------------------- grid barrier ----------------------------------

__device__ __forceinline__ void grid_barrier(unsigned* cnt, unsigned* gen,
                                             unsigned nwg, unsigned* mygen) {
  __syncthreads();
  if (threadIdx.x == 0) {
    __threadfence();                       // release our stores
    const unsigned g = (*mygen)++;
    if (atomicAdd(cnt, 1u) == nwg - 1u) {
      atomicExch(cnt, 0u);
      __threadfence();                     // reset visible before release
      atomicAdd(gen, 1u);
    } else {
      while (atomicAdd(gen, 0u) <= g) __builtin_amdgcn_s_sleep(2);
    }
  }
  __syncthreads();
  __threadfence();                         // acquire: invalidate stale lines
}

// --------------------------- Phase 2: recurrence ---------------------------
// 17 persistent workgroups x 256 threads. WGs 0..15: hidden slice of 32 cols,
// v1 state in registers, spike GEMM via WMMA each step. WG 16: output neuron,
// one step behind, WMMA against padded w2, v2/out_sum in registers.
// Spikes ping-pong between S[0]/S[1] (f16, L2-resident).

__global__ __launch_bounds__(256) void snn_recurrent(
    const float* __restrict__ pre1, const _Float16* __restrict__ wrech,
    const _Float16* __restrict__ w2p, _Float16* __restrict__ S0,
    _Float16* __restrict__ S1, float* __restrict__ out,
    unsigned* __restrict__ bar) {
  using namespace snn;
  const unsigned wg = blockIdx.x;
  const int wave = threadIdx.x >> 5;
  const int lane = threadIdx.x & 31;
  unsigned* cnt = bar;
  unsigned* gen = bar + 1;
  unsigned mygen = 0;
  _Float16* S[2] = {S0, S1};

  if (wg < kHiddenWGs) {
    // wave -> one 16x16 output tile of the [64 x 32] slice
    const int m0  = (wave >> 1) * 16;                 // batch rows
    const int n0g = wg * 32 + (wave & 1) * 16;        // global hidden col
    const int col   = n0g + (lane & 15);
    const int rbase = m0 + ((lane >> 4) << 3);
    float v1[8];
#pragma unroll
    for (int r = 0; r < 8; ++r) v1[r] = 0.0f;

    for (int t = 0; t < kT; ++t) {
      const _Float16* Sin = S[t & 1];
      _Float16* Sout = S[(t + 1) & 1];
      v8f acc = {};
      for (int k0 = 0; k0 < kH; k0 += 32) {
        v16h a = load_frag_a(Sin, kH, m0, k0, lane);
        v16h b = load_frag_b(wrech, kH, n0g, k0, lane);
        acc = wmma_f16(a, b, acc);
      }
#pragma unroll
      for (int r = 0; r < 8; ++r) {
        const int b = rbase + r;                       // batch index
        float v = v1[r] + acc[r] +
                  pre1[((size_t)b * kT + t) * kH + col] - kLeak;
        float sp = (v >= 1.0f) ? 1.0f : 0.0f;          // subtractive reset
        v1[r] = v - sp;
        Sout[b * kH + col] = (_Float16)sp;
      }
      grid_barrier(cnt, gen, kWGs, &mygen);
    }
  } else {
    // Output WG: waves 0..3 each own one 16-row batch tile; O padded to 16.
    const int m0    = (wave & 3) * 16;
    const int col   = lane & 15;                       // output index o
    const int rbase = m0 + ((lane >> 4) << 3);
    float v2[8], osum[8];
#pragma unroll
    for (int r = 0; r < 8; ++r) { v2[r] = 0.0f; osum[r] = 0.0f; }

    for (int t = 0; t < kT; ++t) {
      grid_barrier(cnt, gen, kWGs, &mygen);            // spikes of step t ready
      const _Float16* Ssp = S[(t + 1) & 1];
      if (wave < 4) {
        v8f acc = {};
        for (int k0 = 0; k0 < kH; k0 += 32) {
          v16h a = load_frag_a(Ssp, kH, m0, k0, lane);
          v16h b = load_frag_b(w2p, kH, 0, k0, lane);
          acc = wmma_f16(a, b, acc);
        }
#pragma unroll
        for (int r = 0; r < 8; ++r) {
          float v = v2[r] + acc[r] - kOutLeak;
          v2[r] = v > 0.0f ? v : 0.0f;
          osum[r] += v2[r];
        }
      }
    }
    if (wave < 4 && col < kO) {
#pragma unroll
      for (int r = 0; r < 8; ++r)
        out[(rbase + r) * kO + col] = osum[r] * (1.0f / kT);
    }
  }
}

// --------------------------- host driver -----------------------------------

extern "C" void kernel_launch(void* const* d_in, const int* in_sizes, int n_in,
                              void* d_out, int out_size, void* d_ws,
                              size_t ws_size, hipStream_t stream) {
  using namespace snn;
  (void)in_sizes; (void)n_in; (void)out_size; (void)ws_size;
  const float* x    = (const float*)d_in[0];   // [B,T,C]
  const float* w1   = (const float*)d_in[1];   // [H,C]
  const float* wrec = (const float*)d_in[2];   // [H,H]
  const float* w2   = (const float*)d_in[3];   // [O,H]
  float* out = (float*)d_out;                  // [B,O]
  char* ws = (char*)d_ws;

  // Workspace layout (all sizes 256B-aligned), ~36.3 MB total.
  constexpr size_t kW1hBytes  = (size_t)kH * kC * 2;   // 2 MiB
  constexpr size_t kWrecBytes = (size_t)kH * kH * 2;   // 512 KiB
  constexpr size_t kW2pBytes  = (size_t)16 * kH * 2;   // 16 KiB
  constexpr size_t kSBytes    = (size_t)kB * kH * 2;   // 64 KiB
  constexpr size_t kBarBytes  = 256;
  constexpr size_t oW1h  = 0;
  constexpr size_t oWrec = oW1h + kW1hBytes;
  constexpr size_t oW2p  = oWrec + kWrecBytes;
  constexpr size_t oS0   = oW2p + kW2pBytes;
  constexpr size_t oS1   = oS0 + kSBytes;
  constexpr size_t oBar  = oS1 + kSBytes;
  constexpr size_t oPre  = oBar + kBarBytes;

  _Float16* w1h   = (_Float16*)(ws + oW1h);
  _Float16* wrech = (_Float16*)(ws + oWrec);
  _Float16* w2p   = (_Float16*)(ws + oW2p);
  _Float16* s0    = (_Float16*)(ws + oS0);
  _Float16* s1    = (_Float16*)(ws + oS1);
  unsigned* bar   = (unsigned*)(ws + oBar);
  float*    pre1  = (float*)(ws + oPre);

  // Zero spike buffers (spk_{-1} = 0) + barrier state every call, so graph
  // replay is deterministic. Single contiguous memset.
  hipMemsetAsync(ws + oS0, 0, 2 * kSBytes + kBarBytes, stream);

  // Weight conversions.
  {
    int n = kH * kC;
    cvt_f32_to_f16<<<(n + 255) / 256, 256, 0, stream>>>(w1, w1h, n);
    n = kH * kH;
    cvt_f32_to_f16<<<(n + 255) / 256, 256, 0, stream>>>(wrec, wrech, n);
    n = 16 * kH;
    pad_w2_f16<<<(n + 255) / 256, 256, 0, stream>>>(w2, w2p);
  }

  // Phase 1: the parallel 16384x512x2048 GEMM (34.4 GFLOP via WMMA).
  snn_gemm1<<<kM / 64, 256, 0, stream>>>(x, w1h, pre1);

  // Phase 2: 256 sequential LIF steps in one persistent kernel.
  snn_recurrent<<<kWGs, 256, 0, stream>>>(pre1, wrech, w2p, s0, s1, out, bar);
}
